// RecurrentQ_64733747085607
// MI455X (gfx1250) — compile-verified, matrix operand order checked by reading, not run on hardware
//
#include <hip/hip_runtime.h>
#include <hip/hip_bf16.h>

// ---------------------------------------------------------------------------
// RecurrentQ (encoder MLP -> LSTM scan -> decoder) for gfx1250 (MI455X)
//   * bf16 WMMA (v_wmma_f32_16x16x32_bf16), f32 accumulate
//   * step kernel: W-slab staged in LDS via global_load_async_to_lds_b128
//     (ASYNCcnt), 4-buffer ring with 3 fills in flight; A fragments in a
//     4-slot register ring loaded ahead; B fragments preloaded as a clause
//   * 256 stream-ordered step launches form the scan dependency chain
// ---------------------------------------------------------------------------

typedef unsigned short u16;
typedef unsigned int   u32;
typedef __attribute__((ext_vector_type(16))) __bf16 v16bf;
typedef __attribute__((ext_vector_type(8)))  float  v8f;

namespace cfg {
constexpr int T   = 256;
constexpr int B   = 256;
constexpr int OBS = 256;
constexpr int H   = 1024;
constexpr int L   = 1024;
constexpr int A   = 18;
// LDS staging for k_step: 4 gates x 16 rows x 32 k, rows padded to 40 elems
constexpr int ROWP = 40;                 // u16 elements per padded row
constexpr int SLAB = 4 * 16 * ROWP;      // 2560 u16 = 5120 B per buffer
constexpr int NBUF = 4;                  // async fill ring depth (3 in flight)
}

// -------------------------------- helpers ----------------------------------

__device__ __forceinline__ u16 f32_to_bf16_rne(float f) {
  u32 x = __float_as_uint(f);
  u32 r = x + 0x7FFFu + ((x >> 16) & 1u);   // round-to-nearest-even
  return (u16)(r >> 16);
}

__device__ __forceinline__ float bf16_to_f32(u16 x) {
  return __uint_as_float(((u32)x) << 16);
}

__device__ __forceinline__ float sigmoidf(float x) {
  return 1.0f / (1.0f + __expf(-x));
}

// 16x32 bf16 fragment (A-matrix layout; B from row-major W uses the same
// pattern since B[k][n] = W[n][k]).  Per lane: two contiguous 8-element
// (16-byte) runs at k0 + 8*(lane>>4) and +16.
union FragBF {
  v16bf v;
  uint4 u[2];
};

__device__ __forceinline__ FragBF load_frag(const u16* __restrict__ base,
                                            int row, int ld, int k0, int lane) {
  FragBF f;
  const u16* p = base + (size_t)row * (size_t)ld + (size_t)(k0 + 8 * (lane >> 4));
  f.u[0] = *(const uint4*)(p);
  f.u[1] = *(const uint4*)(p + 16);
  return f;
}

__device__ __forceinline__ v8f wmma_bf16(const FragBF& a, const FragBF& b, v8f c) {
  return __builtin_amdgcn_wmma_f32_16x16x32_bf16(
      /*neg_a=*/false, a.v, /*neg_b=*/false, b.v,
      /*c_mod=*/(short)0, c, /*reuse_a=*/false, /*reuse_b=*/false);
}

// CDNA5 async global->LDS copy (16 B per lane), tracked by ASYNCcnt.
__device__ __forceinline__ void async_copy_b128(u32 lds_byte_addr, const void* gaddr) {
  asm volatile("global_load_async_to_lds_b128 %0, %1, off"
               :: "v"(lds_byte_addr), "v"(gaddr)
               : "memory");
}

__device__ __forceinline__ void wait_async_le2() {
  asm volatile("s_wait_asynccnt 0x2" ::: "memory");
}
__device__ __forceinline__ void wait_async_le1() {
  asm volatile("s_wait_asynccnt 0x1" ::: "memory");
}
__device__ __forceinline__ void wait_async_le0() {
  asm volatile("s_wait_asynccnt 0x0" ::: "memory");
}

// ------------------------------- kernels -----------------------------------

// One LSTM timestep, fully fused:
//   G = enc[t] @ W_ih^T + h_{t-1} @ W_hh^T   (K = H + L = 2048, 64 k-steps)
//   c = sig(f)*c + sig(i)*tanh(g) ; h = sig(o)*tanh(c)
//
// Block = 8 waves sharing one latent tile l0 (16 cols) across 8 batch tiles.
// Per k-step the block async-stages the 4-gate W slab (4x16x32 bf16 = 4 KB,
// one b128 per thread) into a 4-deep LDS ring (3 fills in flight -> each
// fill has ~3 iterations of compute to cover L2 latency); every wave reads
// the same B fragments via a clause of 8 ds_load_b128 (LDS broadcast) into
// registers, then runs the 4-WMMA quartet with staggered dscnt waits.
// A fragments (per-wave batch rows) live in a 4-slot register ring loaded
// 2 iterations ahead.  Accumulators = {i,f,g,o}.
//
// Ring safety (one barrier per iteration, placed between the asynccnt wait
// and the next fill issue):
//   * fill(ks+3) overwrites buf[(ks-1)&3]; its readers (iter ks-1) finished
//     before the barrier at the top of iter ks, and the fill is issued after
//     that barrier.
//   * buf[ks&3] visibility: each thread retires its own fill(ks) with
//     s_wait_asynccnt, then the barrier publishes the whole slab.
__global__ __launch_bounds__(256) void k_step(
    const u16* __restrict__ enc_t,    // [B, H] bf16 (slice of enc)
    const u16* __restrict__ h_prev,   // [B, L] bf16
    const u16* __restrict__ Wih_bf,   // [4L, H]
    const u16* __restrict__ Whh_bf,   // [4L, L]
    const float* __restrict__ b_ih,   // [4L]
    const float* __restrict__ b_hh,   // [4L]
    float* __restrict__ c_f32,        // [B, L] (in/out)
    float* __restrict__ h_f32,        // [B, L] (out, overwritten each step)
    u16* __restrict__ latent_t)       // [B, L] bf16 (= h_t)
{
  using namespace cfg;
  __shared__ u16 sB[NBUF][SLAB];

  const int tid  = threadIdx.x;
  const int lane = tid & 31;
  const int wave = tid >> 5;                 // 0..7
  const int lt   = blockIdx.x & 63;          // latent tile (shared by block)
  const int btg  = blockIdx.x >> 6;          // 0..1
  const int bt   = btg * 8 + wave;           // 0..15 batch tile, per wave
  const int b0   = bt * 16;
  const int l0   = lt * 16;
  const int ar   = lane & 15;

  // staging decomposition: thread -> (gate, row, 16B chunk)
  const int sq = tid >> 6;                   // gate 0..3
  const int sr = (tid >> 2) & 15;            // row 0..15
  const int sc = tid & 3;                    // chunk 0..3 (8 elems each)

  auto stage = [&](int ks) {
    const u16* Wb = (ks < 32) ? Wih_bf : Whh_bf;
    const int  k0 = (ks & 31) * 32;
    const u16* g  = Wb + (size_t)(sq * L + l0 + sr) * 1024 + (size_t)(k0 + sc * 8);
    const u32 lds = (u32)(size_t)(&sB[ks & (NBUF - 1)][(sq * 16 + sr) * ROWP + sc * 8]);
    async_copy_b128(lds, (const void*)g);
  };
  auto loadA = [&](int ks) -> FragBF {
    const u16* Ab = (ks < 32) ? enc_t : h_prev;
    return load_frag(Ab, b0 + ar, 1024, (ks & 31) * 32, lane);
  };

  v8f acc[4] = {v8f{}, v8f{}, v8f{}, v8f{}};   // i, f, g, o

  // prologue: 3 async fills in flight; A fragments for ks=0,1
  stage(0); stage(1); stage(2);
  FragBF aF[4];
  aF[0] = loadA(0);
  aF[1] = loadA(1);
  aF[2] = aF[0];
  aF[3] = aF[0];

  #pragma unroll 4
  for (int ks = 0; ks < 64; ++ks) {
    const int cur = ks & (NBUF - 1);
    // retire fill(ks): 3 outstanding until the tail
    if (ks <= 61)      wait_async_le2();
    else if (ks == 62) wait_async_le1();
    else               wait_async_le0();
    __syncthreads();                          // publish buf[cur] to all waves
    if (ks + 3 < 64) stage(ks + 3);           // refill buf[(ks-1)&3]
    if (ks + 2 < 64) aF[(ks + 2) & 3] = loadA(ks + 2);

    // clause of 8 ds_load_b128 into registers, then the WMMA quartet
    FragBF bq[4];
    #pragma unroll
    for (int q = 0; q < 4; ++q) {
      const u16* p = &sB[cur][(q * 16 + ar) * ROWP + 8 * (lane >> 4)];
      bq[q].u[0] = *(const uint4*)(p);
      bq[q].u[1] = *(const uint4*)(p + 16);
    }
    #pragma unroll
    for (int q = 0; q < 4; ++q) {
      acc[q] = wmma_bf16(aF[ks & 3], bq[q], acc[q]);
    }
  }

  // elementwise LSTM update on this wave's 16x16 (b,l) patch
  const int n = l0 + (lane & 15);
  const float bi = b_ih[0 * L + n] + b_hh[0 * L + n];
  const float bf = b_ih[1 * L + n] + b_hh[1 * L + n];
  const float bg = b_ih[2 * L + n] + b_hh[2 * L + n];
  const float bo = b_ih[3 * L + n] + b_hh[3 * L + n];
  const int mbase = b0 + 8 * (lane >> 4);

  #pragma unroll
  for (int r = 0; r < 8; ++r) {
    const size_t idx = (size_t)(mbase + r) * L + n;
    const float gi = acc[0][r] + bi;
    const float gf = acc[1][r] + bf;
    const float gg = acc[2][r] + bg;
    const float go = acc[3][r] + bo;
    const float c_old = c_f32[idx];
    const float c_new = sigmoidf(gf) * c_old + sigmoidf(gi) * tanhf(gg);
    const float h_new = sigmoidf(go) * tanhf(c_new);
    c_f32[idx]    = c_new;
    h_f32[idx]    = h_new;
    latent_t[idx] = f32_to_bf16_rne(h_new);
  }
}

// fp32 -> bf16 grid-stride convert
__global__ void k_cvt_bf16(const float* __restrict__ in, u16* __restrict__ out, long n) {
  long i = (long)blockIdx.x * blockDim.x + threadIdx.x;
  long stride = (long)gridDim.x * blockDim.x;
  for (; i < n; i += stride) out[i] = f32_to_bf16_rne(in[i]);
}

// zero h0 (bf16) and c (f32)
__global__ void k_zero_state(u16* __restrict__ h0, float* __restrict__ c, int n) {
  int i = blockIdx.x * blockDim.x + threadIdx.x;
  if (i < n) { h0[i] = 0; c[i] = 0.0f; }
}

// enc = relu(state @ W_enc^T + b_enc)  ->  bf16 [T*B, H]
// one wave per 16x64 output tile (4 accumulators sharing one A fragment);
// 65536 wave-tiles -> latency hidden by occupancy.
__global__ __launch_bounds__(256) void k_encoder(
    const u16* __restrict__ state_bf,   // [T*B, OBS]
    const u16* __restrict__ Wenc_bf,    // [H, OBS]
    const float* __restrict__ b_enc,    // [H]
    u16* __restrict__ enc_bf)           // [T*B, H]
{
  using namespace cfg;
  const int lane = threadIdx.x & 31;
  const int wid  = (blockIdx.x * blockDim.x + threadIdx.x) >> 5;
  const int nt = wid & 15;            // 16 tiles of 64 cols over H=1024
  const int mt = wid >> 4;            // 4096 row tiles over T*B=65536
  const int m0 = mt * 16;
  const int n0 = nt * 64;
  const int ar = lane & 15;

  v8f acc[4] = {v8f{}, v8f{}, v8f{}, v8f{}};
  #pragma unroll 2
  for (int ks = 0; ks < OBS / 32; ++ks) {
    const int k0 = ks * 32;
    FragBF a = load_frag(state_bf, m0 + ar, OBS, k0, lane);
    #pragma unroll
    for (int s = 0; s < 4; ++s) {
      FragBF b = load_frag(Wenc_bf, n0 + s * 16 + ar, OBS, k0, lane);
      acc[s] = wmma_bf16(a, b, acc[s]);
    }
  }

  const int mrow = m0 + 8 * (lane >> 4);
  #pragma unroll
  for (int s = 0; s < 4; ++s) {
    const int ncol = n0 + s * 16 + (lane & 15);
    const float bias = b_enc[ncol];
    #pragma unroll
    for (int r = 0; r < 8; ++r) {
      float v = acc[s][r] + bias;
      v = v > 0.0f ? v : 0.0f;
      enc_bf[(size_t)(mrow + r) * H + ncol] = f32_to_bf16_rne(v);
    }
  }
}

// values[t,b,a] = (latent[t,b,:] . W_dec[a,:] + b_dec[a]) * mask[t,b]
__global__ __launch_bounds__(256) void k_decoder(
    const u16* __restrict__ latent,   // [T*B, L] bf16
    const float* __restrict__ W_dec,  // [A, L]
    const float* __restrict__ b_dec,  // [A]
    const float* __restrict__ mask,   // [T*B]
    float* __restrict__ out)          // [T*B, A]
{
  using namespace cfg;
  const int idx = blockIdx.x * blockDim.x + threadIdx.x;
  if (idx >= T * B * A) return;
  const int a  = idx % A;
  const int tb = idx / A;
  const u16*  lrow = latent + (size_t)tb * L;
  const float* wrow = W_dec + (size_t)a * L;
  float acc = 0.0f;
  for (int k = 0; k < L; k += 8) {
    uint4 lv = *(const uint4*)(lrow + k);
    const u32 w[4] = {lv.x, lv.y, lv.z, lv.w};
    #pragma unroll
    for (int j = 0; j < 4; ++j) {
      acc += bf16_to_f32((u16)(w[j] & 0xFFFFu)) * wrow[k + 2 * j + 0];
      acc += bf16_to_f32((u16)(w[j] >> 16))     * wrow[k + 2 * j + 1];
    }
  }
  out[idx] = (acc + b_dec[a]) * mask[tb];
}

// ------------------------------ launcher -----------------------------------

extern "C" void kernel_launch(void* const* d_in, const int* in_sizes, int n_in,
                              void* d_out, int out_size, void* d_ws, size_t ws_size,
                              hipStream_t stream) {
  using namespace cfg;
  (void)in_sizes; (void)n_in; (void)out_size; (void)ws_size;

  const float* state = (const float*)d_in[0];  // [T,B,OBS]
  const float* mask  = (const float*)d_in[1];  // [T,B]
  const float* W_enc = (const float*)d_in[2];  // [H,OBS]
  const float* b_enc = (const float*)d_in[3];  // [H]
  const float* W_ih  = (const float*)d_in[4];  // [4L,H]
  const float* b_ih  = (const float*)d_in[5];  // [4L]
  const float* W_hh  = (const float*)d_in[6];  // [4L,L]
  const float* b_hh  = (const float*)d_in[7];  // [4L]
  const float* W_dec = (const float*)d_in[8];  // [A,L]
  const float* b_dec = (const float*)d_in[9];  // [A]
  float* out = (float*)d_out;                  // values | h | c

  // ---- carve workspace (~322 MB) ----
  char* ws = (char*)d_ws;
  size_t off = 0;
  auto carve = [&](size_t bytes) -> char* {
    char* p = ws + off;
    off = (off + bytes + 255) & ~(size_t)255;
    return p;
  };
  u16*   state_bf  = (u16*)  carve((size_t)T * B * OBS * sizeof(u16));   //  33.6 MB
  u16*   Wenc_bf   = (u16*)  carve((size_t)H * OBS * sizeof(u16));       //   0.5 MB
  u16*   Wih_bf    = (u16*)  carve((size_t)4 * L * H * sizeof(u16));     //   8.4 MB
  u16*   Whh_bf    = (u16*)  carve((size_t)4 * L * L * sizeof(u16));     //   8.4 MB
  u16*   enc_bf    = (u16*)  carve((size_t)T * B * H * sizeof(u16));     // 134.2 MB
  u16*   latent_bf = (u16*)  carve((size_t)T * B * L * sizeof(u16));     // 134.2 MB
  u16*   h0_bf     = (u16*)  carve((size_t)B * L * sizeof(u16));         //   0.5 MB
  float* c_f32     = (float*)carve((size_t)B * L * sizeof(float));       //   1.0 MB
  float* h_f32     = (float*)carve((size_t)B * L * sizeof(float));       //   1.0 MB

  // ---- precision conversion (fp32 -> bf16, RNE) ----
  k_cvt_bf16<<<2048, 256, 0, stream>>>(state, state_bf, (long)T * B * OBS);
  k_cvt_bf16<<<256,  256, 0, stream>>>(W_enc, Wenc_bf, (long)H * OBS);
  k_cvt_bf16<<<1024, 256, 0, stream>>>(W_ih,  Wih_bf,  (long)4 * L * H);
  k_cvt_bf16<<<1024, 256, 0, stream>>>(W_hh,  Whh_bf,  (long)4 * L * L);
  k_zero_state<<<(B * L + 255) / 256, 256, 0, stream>>>(h0_bf, c_f32, B * L);

  // ---- encoder GEMM: 65536 wave-tiles (16x64), 8 waves/block ----
  k_encoder<<<8192, 256, 0, stream>>>(state_bf, Wenc_bf, b_enc, enc_bf);

  // ---- LSTM scan: 256 stream-ordered fused step kernels ----
  // block mapping: blockIdx = btg*64 + lt  (8 waves/block = 8 batch tiles)
  for (int t = 0; t < T; ++t) {
    const u16* enc_t  = enc_bf + (size_t)t * B * H;
    const u16* h_prev = (t == 0) ? h0_bf : (latent_bf + (size_t)(t - 1) * B * L);
    u16* latent_t     = latent_bf + (size_t)t * B * L;
    k_step<<<128, 256, 0, stream>>>(enc_t, h_prev, Wih_bf, Whh_bf,
                                    b_ih, b_hh, c_f32, h_f32, latent_t);
  }

  // ---- decoder + mask ----
  k_decoder<<<(T * B * A + 255) / 256, 256, 0, stream>>>(
      latent_bf, W_dec, b_dec, mask, out);

  // ---- final h, c (fp32) appended after values ----
  hipMemcpyAsync(out + (size_t)T * B * A,           h_f32,
                 (size_t)B * L * sizeof(float), hipMemcpyDeviceToDevice, stream);
  hipMemcpyAsync(out + (size_t)T * B * A + B * L,   c_f32,
                 (size_t)B * L * sizeof(float), hipMemcpyDeviceToDevice, stream);
}
